// SymGNNLayer_20486994002061
// MI455X (gfx1250) — matrix operand order repaired and checked by reading, not compiled, not run on hardware
//
#include <hip/hip_runtime.h>
#include <hip/hip_bf16.h>

typedef __attribute__((ext_vector_type(16))) _Float16 v16h;
typedef __attribute__((ext_vector_type(8)))  float    v8f;

#define NEG_SLOPE 0.01f
#define LN_EPS 1e-5f

// ---------------- WMMA fragment helpers (CDNA5 16x16x32 f16 layouts) ----------------
// A (16x32 f16): lane&15 = row M; lanes 0-15 hold K 0..7,16..23; lanes 16-31 hold K 8..15,24..31.
// Per lane this is two contiguous 8-half runs -> 2x ds_load_b128.
__device__ __forceinline__ v16h a_frag_lds(const _Float16* lds, int lane, int stride, int k0) {
  v16h a;
  const _Float16* p = lds + (size_t)(lane & 15) * stride + k0;
  const int hi = (lane >> 4) << 3;
#pragma unroll
  for (int j = 0; j < 16; ++j) {
    const int i = j >> 1;
    const int k = ((i >> 2) << 4) + hi + ((i & 3) << 1) + (j & 1);
    a[j] = p[k];
  }
  return a;
}

// B (32x16 f16): lane&15 = output column N; lanes 0-15 hold K 0..7,16..23; lanes 16-31 hold K 8..15,24..31.
// Weights pre-staged TRANSPOSED + K-padded: Wt[k][o] = W[o][k]; a lane's 16 halves are strided by Out.
__device__ __forceinline__ v16h b_frag_t(const _Float16* Wt, int lane, int Out, int colbase, int k0) {
  v16h b;
  const _Float16* p = Wt + (size_t)k0 * Out + colbase + (lane & 15);
  const int hi = (lane >> 4) << 3;
#pragma unroll
  for (int j = 0; j < 16; ++j) {
    const int k = ((j >> 3) << 4) + hi + (j & 7);
    b[j] = p[(size_t)k * Out];
  }
  return b;
}

// ordered-int encoding for float atomicMax
__device__ __forceinline__ int   f_enc(float f) { int i = __float_as_int(f); return i >= 0 ? i : (i ^ 0x7fffffff); }
__device__ __forceinline__ float f_dec(int i)   { return __int_as_float(i >= 0 ? i : (i ^ 0x7fffffff)); }

// ---------------- utility kernels ----------------
// d[k*Out + o] = (k < In) ? (f16)s[o*In + k] : 0   (transpose + K-pad)
__global__ void cvt_transpose_pad(const float* __restrict__ s, _Float16* __restrict__ d,
                                  int Out, int In, int InPad) {
  int i = blockIdx.x * 256 + threadIdx.x;
  if (i >= InPad * Out) return;
  const int k = i / Out, o = i - k * Out;
  d[i] = (k < In) ? (_Float16)s[(size_t)o * In + k] : (_Float16)0.f;
}

__global__ void init_bufs(int* __restrict__ maxbuf, float* __restrict__ denom,
                          float* __restrict__ h_out, int n4, size_t n256) {
  size_t i = (size_t)blockIdx.x * blockDim.x + threadIdx.x;
  size_t stride = (size_t)gridDim.x * blockDim.x;
  for (size_t k = i; k < n256; k += stride) h_out[k] = 0.f;
  for (size_t k = i; k < (size_t)n4; k += stride) { maxbuf[k] = (int)0x807fffff; denom[k] = 0.f; }
}

// ---------------- node-side GEMMs: f_ni [N,64], f_nj [N,64], h_src [N,256] ----------------
__device__ __forceinline__ void gemm_tile_k64(const v16h& a0, const v16h& a1, const _Float16* Wt, int Out,
                                              float* out, int ldo, int colbase, int row0, int rlim, int lane) {
  v16h b0 = b_frag_t(Wt, lane, Out, colbase, 0);
  v16h b1 = b_frag_t(Wt, lane, Out, colbase, 32);
  v8f c = {};
  c = __builtin_amdgcn_wmma_f32_16x16x32_f16(false, a0, false, b0, (short)0, c, false, false);
  c = __builtin_amdgcn_wmma_f32_16x16x32_f16(false, a1, false, b1, (short)0, c, false, false);
  const int col = colbase + (lane & 15);
  const int rb = (lane >> 4) << 3;
  float* p = out + (size_t)(row0 + rb) * ldo + col;
  if (row0 + 16 <= rlim) {                      // fast path: whole tile in range
#pragma unroll
    for (int r = 0; r < 8; ++r) p[(size_t)r * ldo] = c[r];
  } else {
#pragma unroll
    for (int r = 0; r < 8; ++r)
      if (row0 + rb + r < rlim) p[(size_t)r * ldo] = c[r];
  }
}

__global__ void node_gemm_kernel(const float* __restrict__ nfeats,
                                 const _Float16* __restrict__ WniT, const _Float16* __restrict__ WnjT,
                                 const _Float16* __restrict__ WsrcT,
                                 float* __restrict__ f_ni, float* __restrict__ f_nj,
                                 float* __restrict__ h_src, int N) {
  __shared__ _Float16 xt[16 * 64];
  const int row0 = blockIdx.x * 16;
  for (int i = threadIdx.x; i < 16 * 64; i += 256) {
    const int r = i >> 6, c = i & 63;
    float v = (row0 + r < N) ? nfeats[(size_t)(row0 + r) * 64 + c] : 0.f;
    xt[i] = (_Float16)v;
  }
  __syncthreads();
  const int lane = threadIdx.x & 31, w = threadIdx.x >> 5;   // 8 waves
  v16h a0 = a_frag_lds(xt, lane, 64, 0);
  v16h a1 = a_frag_lds(xt, lane, 64, 32);
  if (w < 4) gemm_tile_k64(a0, a1, WniT, 64, f_ni, 64, w * 16, row0, N, lane);
  else       gemm_tile_k64(a0, a1, WnjT, 64, f_nj, 64, (w - 4) * 16, row0, N, lane);
  gemm_tile_k64(a0, a1, WsrcT, 256, h_src, 256, w * 16,       row0, N, lane);
  gemm_tile_k64(a0, a1, WsrcT, 256, h_src, 256, (w + 8) * 16, row0, N, lane);
}

// ---- edge GEMM + gather + bias + leaky-relu -> f_out [E,64], fused attention logits [E,4] ----
__global__ void edge_gemm_kernel(const float* __restrict__ efeats,
                                 const int* __restrict__ srcI, const int* __restrict__ dstI,
                                 const _Float16* __restrict__ WfijT,   // [32 (K pad)][64]
                                 const float* __restrict__ f_ni, const float* __restrict__ f_nj,
                                 const float* __restrict__ bias_e, const float* __restrict__ attn,
                                 float* __restrict__ f_out, float* __restrict__ logits, int E) {
  __shared__ _Float16 et[8 * 16 * 32];                      // per-wave 16x32 (K padded 16->32)
  const int w = threadIdx.x >> 5, lane = threadIdx.x & 31;
  const size_t e0 = ((size_t)blockIdx.x * 8 + w) * 16;
  _Float16* myt = et + w * 512;
  for (int i = lane; i < 16 * 32; i += 32) {
    const int r = i >> 5, c = i & 31;
    float v = 0.f;
    if (c < 16 && e0 + r < (size_t)E) v = efeats[(e0 + r) * 16 + c];
    myt[i] = (_Float16)v;
  }
  __syncthreads();
  v16h a = a_frag_lds(myt, lane, 32, 0);
  const int col16 = lane & 15;
  const int rb = (lane >> 4) << 3;
  const bool full = (e0 + 16 <= (size_t)E);

  int se[8], de[8];
#pragma unroll
  for (int r = 0; r < 8; ++r) {
    const size_t e = e0 + rb + r;
    se[r] = (full || e < (size_t)E) ? srcI[e] : 0;
    de[r] = (full || e < (size_t)E) ? dstI[e] : 0;
  }

  for (int t = 0; t < 4; ++t) {                             // 4 head tiles of 16 columns
    v16h b = b_frag_t(WfijT, lane, 64, t * 16, 0);          // K rows 16..31 are staged zeros
    v8f c = {};
    c = __builtin_amdgcn_wmma_f32_16x16x32_f16(false, a, false, b, (short)0, c, false, false);
    const int col = t * 16 + col16;
    const float bcol = bias_e[col];
    const float acol = attn[col];                           // attn[h=t][d=col16]
    float acc[8];
#pragma unroll
    for (int r = 0; r < 8; ++r) {
      const size_t e = e0 + rb + r;
      const bool ok = full || e < (size_t)E;
      float v = c[r] + f_ni[(size_t)se[r] * 64 + col] + f_nj[(size_t)de[r] * 64 + col] + bcol;
      v = v > 0.f ? v : NEG_SLOPE * v;
      if (ok) f_out[e * 64 + col] = v;
      acc[r] = ok ? v * acol : 0.f;
    }
    // fused logit: butterfly-reduce over the 16 lanes of each half-wave (columns of head t)
#pragma unroll
    for (int m = 1; m <= 8; m <<= 1) {
#pragma unroll
      for (int r = 0; r < 8; ++r) acc[r] += __shfl_xor(acc[r], m, 32);
    }
    if (col16 == 0) {
#pragma unroll
      for (int r = 0; r < 8; ++r) {
        const size_t e = e0 + rb + r;
        if (full || e < (size_t)E) logits[e * 4 + t] = acc[r];
      }
    }
  }
}

// ---------------- edge softmax: segment max / exp+sum ----------------
__global__ void segmax_kernel(const float* __restrict__ logits, const int* __restrict__ dstI,
                              int* __restrict__ maxbuf, int E) {
  int idx = blockIdx.x * 256 + threadIdx.x;
  if (idx >= E * 4) return;
  const int e = idx >> 2, h = idx & 3;
  atomicMax(&maxbuf[(size_t)dstI[e] * 4 + h], f_enc(logits[idx]));
}

__global__ void expsum_kernel(float* __restrict__ logits, const int* __restrict__ dstI,
                              const int* __restrict__ maxbuf, float* __restrict__ denom, int E) {
  int idx = blockIdx.x * 256 + threadIdx.x;
  if (idx >= E * 4) return;
  const int e = idx >> 2, h = idx & 3;
  const size_t di = (size_t)dstI[e] * 4 + h;
  const float ex = __expf(logits[idx] - f_dec(maxbuf[di]));
  logits[idx] = ex;
  atomicAdd(&denom[di], ex);
}

// ---------------- weighted aggregation into h_out [N,4,64] ----------------
__global__ void aggregate_kernel(const float* __restrict__ h_src,
                                 const int* __restrict__ srcI, const int* __restrict__ dstI,
                                 const float* __restrict__ exv, const float* __restrict__ denom,
                                 float* __restrict__ h_out, int E) {
  int idx = blockIdx.x * 256 + threadIdx.x;
  if (idx >= E * 4) return;
  const int e = idx >> 2, h = idx & 3;
  const int d = dstI[e];
  const float a = exv[idx] / denom[(size_t)d * 4 + h];
  const float4* sp = (const float4*)(h_src + (size_t)srcI[e] * 256 + h * 64);
  float* op = h_out + (size_t)d * 256 + h * 64;
  __builtin_prefetch(sp, 0, 1);
#pragma unroll 4
  for (int k = 0; k < 16; ++k) {
    float4 v = sp[k];
    atomicAdd(&op[4 * k + 0], v.x * a);
    atomicAdd(&op[4 * k + 1], v.y * a);
    atomicAdd(&op[4 * k + 2], v.z * a);
    atomicAdd(&op[4 * k + 3], v.w * a);
  }
}

// ---------------- node LN + projection (256->64) + residual ----------------
__global__ void node_final_kernel(const float* __restrict__ h_out, const _Float16* __restrict__ WpnT,
                                  const float* __restrict__ g, const float* __restrict__ bln,
                                  const float* __restrict__ bp, const float* __restrict__ nfeats,
                                  float* __restrict__ out, int N) {
  __shared__ _Float16 xt[16 * 256];
  const int row0 = blockIdx.x * 16;
  const int tid = threadIdx.x;                               // 128 threads = 4 waves
  if (tid < 16) {
    const int row = row0 + tid;
    if (row < N) {
      const float4* p4 = (const float4*)(h_out + (size_t)row * 256);
      float s = 0.f, ss = 0.f;
      for (int k = 0; k < 64; ++k) {
        float4 v = p4[k];
        s  += v.x + v.y + v.z + v.w;
        ss += v.x * v.x + v.y * v.y + v.z * v.z + v.w * v.w;
      }
      const float mean = s * (1.f / 256.f);
      const float inv = rsqrtf(fmaxf(ss * (1.f / 256.f) - mean * mean, 0.f) + LN_EPS);
      const float* p = h_out + (size_t)row * 256;
      for (int k = 0; k < 256; ++k)
        xt[tid * 256 + k] = (_Float16)((p[k] - mean) * inv * g[k] + bln[k]);
    } else {
      for (int k = 0; k < 256; ++k) xt[tid * 256 + k] = (_Float16)0.f;
    }
  }
  __syncthreads();
  const int lane = tid & 31, w = tid >> 5;
  const int colbase = w * 16;
  v8f c = {};
  for (int k0 = 0; k0 < 256; k0 += 32) {
    v16h a = a_frag_lds(xt, lane, 256, k0);
    v16h b = b_frag_t(WpnT, lane, 64, colbase, k0);
    c = __builtin_amdgcn_wmma_f32_16x16x32_f16(false, a, false, b, (short)0, c, false, false);
  }
  const int col = colbase + (lane & 15);
  const int rb = (lane >> 4) << 3;
  float* p = out + (size_t)(row0 + rb) * 64 + col;
  const float* nf = nfeats + (size_t)(row0 + rb) * 64 + col;
  const float bc = bp[col];
  if (row0 + 16 <= N) {
#pragma unroll
    for (int r = 0; r < 8; ++r) p[(size_t)r * 64] = c[r] + bc + nf[(size_t)r * 64];
  } else {
#pragma unroll
    for (int r = 0; r < 8; ++r)
      if (row0 + rb + r < N) p[(size_t)r * 64] = c[r] + bc + nf[(size_t)r * 64];
  }
}

// ---------------- edge LN + projection (64->16) + residual ----------------
__global__ void edge_final_kernel(const float* __restrict__ f_out, const _Float16* __restrict__ WpeT,
                                  const float* __restrict__ g, const float* __restrict__ bln,
                                  const float* __restrict__ bp, const float* __restrict__ efeats,
                                  float* __restrict__ out_e, int E) {
  __shared__ _Float16 et[8 * 16 * 64];
  const int w = threadIdx.x >> 5, lane = threadIdx.x & 31;
  const size_t e0 = ((size_t)blockIdx.x * 8 + w) * 16;
  _Float16* myt = et + w * 1024;
  if (lane < 16) {
    const size_t e = e0 + lane;
    if (e < (size_t)E) {
      const float4* p4 = (const float4*)(f_out + e * 64);
      float s = 0.f, ss = 0.f;
      for (int k = 0; k < 16; ++k) {
        float4 v = p4[k];
        s  += v.x + v.y + v.z + v.w;
        ss += v.x * v.x + v.y * v.y + v.z * v.z + v.w * v.w;
      }
      const float mean = s * (1.f / 64.f);
      const float inv = rsqrtf(fmaxf(ss * (1.f / 64.f) - mean * mean, 0.f) + LN_EPS);
      const float* p = f_out + e * 64;
      for (int k = 0; k < 64; ++k)
        myt[lane * 64 + k] = (_Float16)((p[k] - mean) * inv * g[k] + bln[k]);
    } else {
      for (int k = 0; k < 64; ++k) myt[lane * 64 + k] = (_Float16)0.f;
    }
  }
  __syncthreads();
  v8f c = {};
  for (int k0 = 0; k0 < 64; k0 += 32) {
    v16h a = a_frag_lds(myt, lane, 64, k0);
    v16h b = b_frag_t(WpeT, lane, 16, 0, k0);
    c = __builtin_amdgcn_wmma_f32_16x16x32_f16(false, a, false, b, (short)0, c, false, false);
  }
  const int col = lane & 15;
  const int rb = (lane >> 4) << 3;
  if (e0 + 16 <= (size_t)E) {
#pragma unroll
    for (int r = 0; r < 8; ++r) {
      const size_t e = e0 + rb + r;
      out_e[e * 16 + col] = c[r] + bp[col] + efeats[e * 16 + col];
    }
  } else {
#pragma unroll
    for (int r = 0; r < 8; ++r) {
      const size_t e = e0 + rb + r;
      if (e < (size_t)E) out_e[e * 16 + col] = c[r] + bp[col] + efeats[e * 16 + col];
    }
  }
}

// ---------------- host-side launch ----------------
extern "C" void kernel_launch(void* const* d_in, const int* in_sizes, int n_in,
                              void* d_out, int out_size, void* d_ws, size_t ws_size,
                              hipStream_t stream) {
  const float* nfeats = (const float*)d_in[0];
  const float* efeats = (const float*)d_in[1];
  const int*   srcI   = (const int*)d_in[2];
  const int*   dstI   = (const int*)d_in[3];
  const float* W_ni   = (const float*)d_in[4];
  const float* W_nj   = (const float*)d_in[5];
  const float* W_fij  = (const float*)d_in[6];
  const float* W_src  = (const float*)d_in[7];
  const float* attn   = (const float*)d_in[8];
  const float* bias_e = (const float*)d_in[9];
  const float* ln_n_g = (const float*)d_in[10];
  const float* ln_n_b = (const float*)d_in[11];
  const float* ln_e_g = (const float*)d_in[12];
  const float* ln_e_b = (const float*)d_in[13];
  const float* Wp_n   = (const float*)d_in[14];
  const float* bp_n   = (const float*)d_in[15];
  const float* Wp_e   = (const float*)d_in[16];
  const float* bp_e   = (const float*)d_in[17];

  const int N = in_sizes[0] / 64;
  const int E = in_sizes[1] / 16;

  // workspace layout (floats)
  float* ws = (float*)d_ws;
  size_t off = 0;
  float* f_ni  = ws + off; off += (size_t)N * 64;
  float* f_nj  = ws + off; off += (size_t)N * 64;
  float* h_src = ws + off; off += (size_t)N * 256;
  float* h_out = ws + off; off += (size_t)N * 256;
  float* f_out = ws + off; off += (size_t)E * 64;
  float* exv   = ws + off; off += (size_t)E * 4;   // logits, then exp values
  int*   maxbuf = (int*)(ws + off); off += (size_t)N * 4;
  float* denom = ws + off; off += (size_t)N * 4;
  // transposed+padded f16 weights: Wt[kPad][Out]
  _Float16* WniT  = (_Float16*)(ws + off);
  _Float16* WnjT  = WniT  + 64 * 64;     // [64][64]
  _Float16* WfijT = WnjT  + 64 * 64;     // [32][64] (K 16 padded to 32)
  _Float16* WsrcT = WfijT + 32 * 64;     // [64][256]
  _Float16* WpnT  = WsrcT + 64 * 256;    // [256][64]
  _Float16* WpeT  = WpnT  + 256 * 64;    // [64][16]

  // 1. weight transpose+pad conversion to f16
  cvt_transpose_pad<<<(64 * 64 + 255) / 256, 256, 0, stream>>>(W_ni,  WniT,  64, 64, 64);
  cvt_transpose_pad<<<(64 * 64 + 255) / 256, 256, 0, stream>>>(W_nj,  WnjT,  64, 64, 64);
  cvt_transpose_pad<<<(32 * 64 + 255) / 256, 256, 0, stream>>>(W_fij, WfijT, 64, 16, 32);
  cvt_transpose_pad<<<(64 * 256 + 255) / 256, 256, 0, stream>>>(W_src, WsrcT, 256, 64, 64);
  cvt_transpose_pad<<<(256 * 64 + 255) / 256, 256, 0, stream>>>(Wp_n,  WpnT,  64, 256, 256);
  cvt_transpose_pad<<<(64 * 16 + 255) / 256, 256, 0, stream>>>(Wp_e,  WpeT,  16, 64, 64);

  // 2. accumulator / softmax-state init
  init_bufs<<<2048, 256, 0, stream>>>(maxbuf, denom, h_out, N * 4, (size_t)N * 256);

  // 3. dense node GEMMs (WMMA)
  node_gemm_kernel<<<(N + 15) / 16, 256, 0, stream>>>(nfeats, WniT, WnjT, WsrcT,
                                                      f_ni, f_nj, h_src, N);
  // 4. edge GEMM + gather + leaky-relu + fused logits (WMMA + lane butterflies)
  edge_gemm_kernel<<<(E + 127) / 128, 256, 0, stream>>>(efeats, srcI, dstI, WfijT,
                                                        f_ni, f_nj, bias_e, attn,
                                                        f_out, exv, E);
  // 5-7. edge softmax + aggregation
  const int ehBlocks = (E * 4 + 255) / 256;
  segmax_kernel<<<ehBlocks, 256, 0, stream>>>(exv, dstI, maxbuf, E);
  expsum_kernel<<<ehBlocks, 256, 0, stream>>>(exv, dstI, maxbuf, denom, E);
  aggregate_kernel<<<ehBlocks, 256, 0, stream>>>(h_src, srcI, dstI, exv, denom, h_out, E);

  // 8-9. layernorm + projection + residual (WMMA)
  float* out_n = (float*)d_out;
  float* out_e = out_n + (size_t)N * 64;
  node_final_kernel<<<(N + 15) / 16, 128, 0, stream>>>(h_out, WpnT, ln_n_g, ln_n_b,
                                                       bp_n, nfeats, out_n, N);
  edge_final_kernel<<<(E + 127) / 128, 256, 0, stream>>>(f_out, WpeT, ln_e_g, ln_e_b,
                                                         bp_e, efeats, out_e, E);
}
